// AoA_Refine_Module_50586124812561
// MI455X (gfx1250) — matrix operand order repaired and checked by reading, not compiled, not run on hardware
//
#include <hip/hip_runtime.h>

// ---------------- problem constants ----------------
#define B_   32
#define M_   196
#define D_   1024
#define H_   8
#define HD_  128
#define L_   3
#define MP_  224              // M padded to multiple of 32 for WMMA K-loops
#define BM_  (B_ * M_)        // 6272 (= 98 * 64)
#define BH_  (B_ * H_)        // 256

typedef __attribute__((ext_vector_type(16))) __bf16 v16bf;
typedef __attribute__((ext_vector_type(8)))  float  v8f;

// ---- CDNA5 async global->LDS path (ASYNCcnt), with graceful fallback ----
#if defined(__has_builtin)
#if __has_builtin(__builtin_amdgcn_global_load_async_to_lds_b128) && \
    __has_builtin(__builtin_amdgcn_s_wait_asynccnt)
#define HAVE_ASYNC_LDS 1
#endif
#endif
#ifndef HAVE_ASYNC_LDS
#define HAVE_ASYNC_LDS 0
#endif

#if HAVE_ASYNC_LDS
typedef int v4i __attribute__((vector_size(16)));
typedef __attribute__((address_space(1))) v4i* gas_v4i_ptr;
typedef __attribute__((address_space(3))) v4i* las_v4i_ptr;
__device__ __forceinline__ void async_ld_b128(const void* g, void* l) {
    __builtin_amdgcn_global_load_async_to_lds_b128(
        (gas_v4i_ptr)(g), (las_v4i_ptr)(l), /*offset=*/0, /*cpol=*/0);
}
#endif

struct ProjPtrs {
    const float* bias[4];
    const float* gw[4];
    const float* gb[4];
    __bf16*      out[4];   // [B,H,MP_,HD_] bf16
    __bf16*      v2T;      // [B,H,HD_,MP_] bf16 (transposed v2)
};

__device__ __forceinline__ v16bf frag16(const __bf16* p0, const __bf16* p1) {
    v16bf r;
    ((uint4*)&r)[0] = *(const uint4*)p0;   // elements 0..7
    ((uint4*)&r)[1] = *(const uint4*)p1;   // elements 8..15
    return r;
}

__device__ __forceinline__ v8f wmma_bf16(v16bf a, v16bf b, v8f c) {
    return __builtin_amdgcn_wmma_f32_16x16x32_bf16(
        /*neg_a=*/false, a, /*neg_b=*/false, b,
        /*c_mod=*/(short)0, c, /*reuse_a=*/false, /*reuse_b=*/false);
}

// ---------------- fp32 -> bf16 convert ----------------
__global__ void k_cvt(const float* __restrict__ s, __bf16* __restrict__ d, int n) {
    int i = blockIdx.x * 256 + threadIdx.x;
    if (i < n) d[i] = (__bf16)s[i];
}

// -------- elementwise bf16 product (v1 * k) --------
__global__ void k_mul_bf16(const __bf16* __restrict__ a, const __bf16* __restrict__ b,
                           __bf16* __restrict__ d, int n) {
    int i = blockIdx.x * 256 + threadIdx.x;
    if (i < n) d[i] = (__bf16)((float)a[i] * (float)b[i]);
}

// ---------------- projection: y = GN_head(CELU(x @ W^T + b)) ----------------
// grid: (BM_/64, H_, 4 projections), block 256 (8 waves).
// Block computes 64 rows x 128 cols (one head) of one projection, K=1024.
// Double-buffered LDS staging via GLOBAL_LOAD_ASYNC_TO_LDS_B128: while the
// waves run 8 WMMAs on buffer[i], the next 24KB tile streams into buffer[i^1]
// with no VGPR involvement (ASYNCcnt-tracked).
// Epilogue Outs[64][128] overlays the (then dead) staging buffers.
__global__ __launch_bounds__(256) void k_proj(const __bf16* __restrict__ xb,   // [BM_, D_]
                                              const __bf16* __restrict__ Wb,   // [4, D_, D_] (row o, col i)
                                              ProjPtrs p) {
    // buffer b: A tile [64][64] at +0 (8KB), B tile [128][64] at +8192 (16KB)
    __shared__ __align__(16) unsigned char smemraw[2 * 24576];   // 48 KB
    __shared__ float smu[64], srs[64];

    const int t    = threadIdx.x;
    const int wave = t >> 5, lane = t & 31, half = lane >> 4, ln = lane & 15;
    const int z = blockIdx.z, h = blockIdx.y;
    const int r0 = blockIdx.x * 64, c0 = h * HD_;
    const __bf16* W = Wb + (size_t)z * D_ * D_;

    v8f acc0 = {}, acc1 = {}, acc2 = {}, acc3 = {};

    auto stage = [&](unsigned char* buf, int k0) {
        __bf16 (*A)[64]  = (__bf16 (*)[64])buf;
        __bf16 (*Bt)[64] = (__bf16 (*)[64])(buf + 8192);
#if HAVE_ASYNC_LDS
        #pragma unroll
        for (int rep = 0; rep < 2; ++rep) {            // A: 512 b128 chunks
            int ch = t + rep * 256;
            int row = ch >> 3, part = ch & 7;
            async_ld_b128(xb + (size_t)(r0 + row) * D_ + k0 + part * 8, &A[row][part * 8]);
        }
        #pragma unroll
        for (int rep = 0; rep < 4; ++rep) {            // B: 1024 b128 chunks
            int ch = t + rep * 256;
            int o = ch >> 3, part = ch & 7;
            async_ld_b128(W + (size_t)(c0 + o) * D_ + k0 + part * 8, &Bt[o][part * 8]);
        }
#else
        // Fallback: batch loads into distinct registers, then store (MLP).
        uint4 ra[2], rb[4];
        #pragma unroll
        for (int rep = 0; rep < 2; ++rep) {
            int ch = t + rep * 256;
            int row = ch >> 3, part = ch & 7;
            ra[rep] = *(const uint4*)(xb + (size_t)(r0 + row) * D_ + k0 + part * 8);
        }
        #pragma unroll
        for (int rep = 0; rep < 4; ++rep) {
            int ch = t + rep * 256;
            int o = ch >> 3, part = ch & 7;
            rb[rep] = *(const uint4*)(W + (size_t)(c0 + o) * D_ + k0 + part * 8);
        }
        #pragma unroll
        for (int rep = 0; rep < 2; ++rep) {
            int ch = t + rep * 256;
            int row = ch >> 3, part = ch & 7;
            *(uint4*)&A[row][part * 8] = ra[rep];
        }
        #pragma unroll
        for (int rep = 0; rep < 4; ++rep) {
            int ch = t + rep * 256;
            int o = ch >> 3, part = ch & 7;
            *(uint4*)&Bt[o][part * 8] = rb[rep];
        }
#endif
    };

    auto compute = [&](const unsigned char* buf) {
        const __bf16 (*A)[64]  = (const __bf16 (*)[64])buf;
        const __bf16 (*Bt)[64] = (const __bf16 (*)[64])(buf + 8192);
        #pragma unroll
        for (int ks = 0; ks < 2; ++ks) {
            const int kb = ks * 32;
            // B frag: ln selects N column, half selects K chunks half*8 and 16+half*8
            v16bf b = frag16(&Bt[wave * 16 + ln][kb + half * 8],
                             &Bt[wave * 16 + ln][kb + 16 + half * 8]);
            // 4 M-subtiles reuse the same B fragment
            v16bf a0 = frag16(&A[ 0 + ln][kb + half * 8], &A[ 0 + ln][kb + 16 + half * 8]);
            acc0 = wmma_bf16(a0, b, acc0);
            v16bf a1 = frag16(&A[16 + ln][kb + half * 8], &A[16 + ln][kb + 16 + half * 8]);
            acc1 = wmma_bf16(a1, b, acc1);
            v16bf a2 = frag16(&A[32 + ln][kb + half * 8], &A[32 + ln][kb + 16 + half * 8]);
            acc2 = wmma_bf16(a2, b, acc2);
            v16bf a3 = frag16(&A[48 + ln][kb + half * 8], &A[48 + ln][kb + 16 + half * 8]);
            acc3 = wmma_bf16(a3, b, acc3);
        }
    };

    unsigned char* base0 = smemraw;
    unsigned char* base1 = smemraw + 24576;

    stage(base0, 0);
#if HAVE_ASYNC_LDS
    __builtin_amdgcn_s_wait_asynccnt(0);
#endif
    __syncthreads();

    for (int kt = 0; kt < D_ / 64; ++kt) {
        unsigned char* cur = (kt & 1) ? base1 : base0;
        unsigned char* nxt = (kt & 1) ? base0 : base1;
        if (kt + 1 < D_ / 64) {
            stage(nxt, (kt + 1) * 64);                 // async: no VGPRs, no wait yet
            if (kt + 2 < D_ / 64)                      // warm L2 one tile further out
                __builtin_prefetch(W + (size_t)(c0 + (t >> 1)) * D_ + (kt + 2) * 64, 0, 1);
        }
        compute(cur);                                  // 8 WMMAs overlap the DMA
#if HAVE_ASYNC_LDS
        __builtin_amdgcn_s_wait_asynccnt(0);
#endif
        __syncthreads();
    }
    // staging buffers are dead from here on (final barrier above): overlay Outs
    float (*Outs)[HD_] = (float (*)[HD_])smemraw;      // 32 KB

    // ---- bias + CELU into LDS ----
    const int colh = wave * 16 + ln;
    const float bv = p.bias[z][c0 + colh];
    v8f accs[4] = {acc0, acc1, acc2, acc3};
    #pragma unroll
    for (int mt = 0; mt < 4; ++mt) {
        #pragma unroll
        for (int v = 0; v < 8; ++v) {
            float y = accs[mt][v] + bv;
            y = (y > 0.f) ? y : 1.3f * (__expf(y * (1.0f / 1.3f)) - 1.0f);
            Outs[mt * 16 + v + 8 * half][colh] = y;
        }
    }
    __syncthreads();

    // ---- per-row stats over the head's 128 columns (4 threads per row) ----
    {
        int row = t >> 2, j = t & 3;
        float s = 0.f, sq = 0.f;
        #pragma unroll
        for (int jj = 0; jj < 32; ++jj) {
            float x = Outs[row][j + 4 * jj];
            s += x; sq += x * x;
        }
        #pragma unroll
        for (int o = 1; o < 4; o <<= 1) {
            s  += __shfl_xor(s,  o, 4);
            sq += __shfl_xor(sq, o, 4);
        }
        if (j == 0) {
            float mu  = s  * (1.0f / 128.0f);
            float var = sq * (1.0f / 128.0f) - mu * mu;
            smu[row] = mu;
            srs[row] = rsqrtf(var + 1e-5f);
        }
    }
    __syncthreads();

    // ---- normalize, scale, write bf16 (+ transposed copy for v2) ----
    const float gwv = p.gw[z][c0 + colh], gbv = p.gb[z][c0 + colh];
    __bf16* outp = p.out[z];
    #pragma unroll
    for (int mt = 0; mt < 4; ++mt) {
        #pragma unroll
        for (int v = 0; v < 8; ++v) {
            int row = mt * 16 + v + 8 * half;
            int r = r0 + row;
            int b = r / M_, m = r - b * M_;
            float val = (Outs[row][colh] - smu[row]) * srs[row] * gwv + gbv;
            size_t bh = (size_t)(b * H_ + h);
            outp[(bh * MP_ + m) * HD_ + colh] = (__bf16)val;
            if (z == 3) p.v2T[(bh * HD_ + colh) * MP_ + m] = (__bf16)val;
        }
    }
}

// ---------------- scores + softmax -> alpha (bf16, K padded to 224) ----------------
// grid: (13 q-tiles, B*H), block 256 (8 waves).
__global__ __launch_bounds__(256) void k_scores(const __bf16* __restrict__ q2,
                                                const __bf16* __restrict__ kmat,
                                                const float*  __restrict__ mask,
                                                __bf16* __restrict__ alpha) {
    __shared__ float Sc[16][MP_];
    const int t = threadIdx.x, wave = t >> 5, lane = t & 31, half = lane >> 4, ln = lane & 15;
    const int bh = blockIdx.y, b = bh >> 3;
    const int q0 = blockIdx.x * 16;
    const size_t baseQ = ((size_t)bh * MP_ + q0 + ln) * HD_;

    for (int kt = wave; kt < MP_ / 16; kt += 8) {
        v8f acc = {};
        const size_t baseK = ((size_t)bh * MP_ + kt * 16 + ln) * HD_;
        #pragma unroll
        for (int c = 0; c < HD_; c += 32) {
            v16bf a  = frag16(q2 + baseQ + c + half * 8, q2 + baseQ + c + 16 + half * 8);
            v16bf bb = frag16(kmat + baseK + c + half * 8, kmat + baseK + c + 16 + half * 8);
            acc = wmma_bf16(a, bb, acc);
        }
        const int col = kt * 16 + ln;
        const float mv = (col < M_) ? mask[b * M_ + col] : 0.f;
        #pragma unroll
        for (int v = 0; v < 8; ++v) {
            float s = acc[v] * 0.08838834764831845f;   // 1/sqrt(128)
            if (mv == 0.f) s = -1.0e9f;
            Sc[v + 8 * half][col] = s;
        }
    }
    __syncthreads();

    const int row = t >> 4, j = t & 15;
    float vmax = -3.0e38f;
    for (int jj = j; jj < MP_; jj += 16) vmax = fmaxf(vmax, Sc[row][jj]);
    #pragma unroll
    for (int o = 1; o < 16; o <<= 1) vmax = fmaxf(vmax, __shfl_xor(vmax, o, 16));
    float ssum = 0.f;
    for (int jj = j; jj < MP_; jj += 16) {
        float e = __expf(Sc[row][jj] - vmax);
        Sc[row][jj] = e; ssum += e;
    }
    #pragma unroll
    for (int o = 1; o < 16; o <<= 1) ssum += __shfl_xor(ssum, o, 16);
    const float inv = 1.f / ssum;
    __bf16* ap = alpha + ((size_t)bh * MP_ + q0 + row) * MP_;
    for (int jj = j; jj < MP_; jj += 16) ap[jj] = (__bf16)(Sc[row][jj] * inv);
}

// ---------------- attn = alpha @ v2 ; gate = sigmoid((v1*k) @ Wc^T + bc) ----------------
// grid: (13 q-tiles, B*H), block 256; wave w owns output cols w*16..w*16+15.
__global__ __launch_bounds__(256) void k_attn(const __bf16* __restrict__ alpha,  // [BH,MP_,MP_]
                                              const __bf16* __restrict__ v2T,    // [BH,HD_,MP_]
                                              const __bf16* __restrict__ v1k,    // [BH,MP_,HD_]
                                              const __bf16* __restrict__ Wcb,    // [HD_,HD_]
                                              const float*  __restrict__ bc,
                                              float* __restrict__ gout) {        // [B,M_,D_]
    const int t = threadIdx.x, wave = t >> 5, lane = t & 31, half = lane >> 4, ln = lane & 15;
    const int bh = blockIdx.y, b = bh >> 3, h = bh & 7;
    const int q0 = blockIdx.x * 16, c0w = wave * 16;

    v8f attn = {};
    {
        const size_t baseA = ((size_t)bh * MP_ + q0 + ln) * MP_;
        const size_t baseB = ((size_t)bh * HD_ + c0w + ln) * MP_;
        #pragma unroll
        for (int kk = 0; kk < MP_; kk += 32) {
            v16bf a  = frag16(alpha + baseA + kk + half * 8, alpha + baseA + kk + 16 + half * 8);
            v16bf bb = frag16(v2T + baseB + kk + half * 8, v2T + baseB + kk + 16 + half * 8);
            attn = wmma_bf16(a, bb, attn);
        }
    }
    v8f g = {};
    {
        const size_t baseA = ((size_t)bh * MP_ + q0 + ln) * HD_;
        const size_t baseB = (size_t)(c0w + ln) * HD_;
        #pragma unroll
        for (int kk = 0; kk < HD_; kk += 32) {
            v16bf a  = frag16(v1k + baseA + kk + half * 8, v1k + baseA + kk + 16 + half * 8);
            v16bf bb = frag16(Wcb + baseB + kk + half * 8, Wcb + baseB + kk + 16 + half * 8);
            g = wmma_bf16(a, bb, g);
        }
    }
    const float bcv = bc[c0w + ln];
    #pragma unroll
    for (int v = 0; v < 8; ++v) {
        int row = v + 8 * half, m = q0 + row;
        if (m < M_) {
            float gate = 1.f / (1.f + __expf(-(g[v] + bcv)));
            gout[((size_t)(b * M_ + m)) * D_ + h * HD_ + c0w + ln] = gate * attn[v];
        }
    }
}

// ---------------- residual + LayerNorm over D ----------------
// grid: BM_ blocks, 256 threads (4 elems each).
__global__ __launch_bounds__(256) void k_resln(float* __restrict__ x,
                                               const float* __restrict__ ga,
                                               const float* __restrict__ lnw,
                                               const float* __restrict__ lnb,
                                               __bf16* __restrict__ xb,
                                               float* __restrict__ xout) {   // may be null
    __shared__ float red[256];
    const int t = threadIdx.x;
    const size_t base = (size_t)blockIdx.x * D_;
    float y[4];
    float s = 0.f;
    #pragma unroll
    for (int kq = 0; kq < 4; ++kq) {
        int d = t + kq * 256;
        y[kq] = x[base + d] + ga[base + d];
        s += y[kq];
    }
    red[t] = s; __syncthreads();
    for (int o = 128; o > 0; o >>= 1) { if (t < o) red[t] += red[t + o]; __syncthreads(); }
    float mu = red[0] * (1.0f / D_);
    __syncthreads();
    float sq = 0.f;
    #pragma unroll
    for (int kq = 0; kq < 4; ++kq) { float dd = y[kq] - mu; sq += dd * dd; }
    red[t] = sq; __syncthreads();
    for (int o = 128; o > 0; o >>= 1) { if (t < o) red[t] += red[t + o]; __syncthreads(); }
    float rs = rsqrtf(red[0] * (1.0f / D_) + 1e-5f);
    #pragma unroll
    for (int kq = 0; kq < 4; ++kq) {
        int d = t + kq * 256;
        float xn = (y[kq] - mu) * rs * lnw[d] + lnb[d];
        x[base + d]  = xn;
        xb[base + d] = (__bf16)xn;
        if (xout) xout[base + d] = xn;
    }
}

// ---------------- masked mean pool ----------------
__global__ void k_pool(const float* __restrict__ x, const float* __restrict__ mask,
                       float* __restrict__ gv) {
    int gid = blockIdx.x * 256 + threadIdx.x;   // B_*D_ threads
    if (gid >= B_ * D_) return;
    int b = gid / D_, d = gid - b * D_;
    float acc = 0.f, ms = 0.f;
    for (int m = 0; m < M_; ++m) {
        float mv = mask[b * M_ + m];
        acc += x[((size_t)(b * M_ + m)) * D_ + d] * mv;
        ms  += mv;
    }
    gv[gid] = acc / ms;
}

// ---------------- host orchestration ----------------
extern "C" void kernel_launch(void* const* d_in, const int* in_sizes, int n_in,
                              void* d_out, int out_size, void* d_ws, size_t ws_size,
                              hipStream_t stream) {
    const float* att_feats = (const float*)d_in[0];
    const float* att_mask  = (const float*)d_in[1];
    const float* Wq2 = (const float*)d_in[2];
    const float* bq2 = (const float*)d_in[3];
    const float* gq2w = (const float*)d_in[4];
    const float* gq2b = (const float*)d_in[5];
    const float* Wk  = (const float*)d_in[6];
    const float* bk  = (const float*)d_in[7];
    const float* gkw = (const float*)d_in[8];
    const float* gkb = (const float*)d_in[9];
    const float* Wv1 = (const float*)d_in[10];
    const float* bv1 = (const float*)d_in[11];
    const float* gv1w = (const float*)d_in[12];
    const float* gv1b = (const float*)d_in[13];
    const float* Wv2 = (const float*)d_in[14];
    const float* bv2 = (const float*)d_in[15];
    const float* gv2w = (const float*)d_in[16];
    const float* gv2b = (const float*)d_in[17];
    const float* Wc  = (const float*)d_in[18];
    const float* bc  = (const float*)d_in[19];
    const float* lnw = (const float*)d_in[20];
    const float* lnb = (const float*)d_in[21];

    // ----- carve workspace -----
    char* pws = (char*)d_ws;
    auto alloc = [&](size_t bytes) -> char* {
        char* r = pws;
        pws += (bytes + 255) & ~(size_t)255;
        return r;
    };
    const size_t projElems = (size_t)BH_ * MP_ * HD_;
    float*  x     = (float*)alloc((size_t)BM_ * D_ * 4);
    __bf16* xb    = (__bf16*)alloc((size_t)BM_ * D_ * 2);
    __bf16* Wb    = (__bf16*)alloc((size_t)4 * D_ * D_ * 2);
    __bf16* Wcb   = (__bf16*)alloc((size_t)HD_ * HD_ * 2);
    __bf16* q2b   = (__bf16*)alloc(projElems * 2);
    __bf16* kb    = (__bf16*)alloc(projElems * 2);
    __bf16* v1b   = (__bf16*)alloc(projElems * 2);
    __bf16* v2b   = (__bf16*)alloc(projElems * 2);
    __bf16* v2T   = (__bf16*)alloc(projElems * 2);
    __bf16* v1k   = (__bf16*)alloc(projElems * 2);
    __bf16* alpha = (__bf16*)alloc((size_t)BH_ * MP_ * MP_ * 2);
    float*  gattn = (float*)alloc((size_t)BM_ * D_ * 4);

    float* gv_out = (float*)d_out;                    // [B, D]
    float* x_out  = (float*)d_out + (size_t)B_ * D_;  // [B, M, D]

    // x = att_feats (f32 working copy), xb = bf16(att_feats)
    (void)hipMemcpyAsync(x, att_feats, (size_t)BM_ * D_ * 4, hipMemcpyDeviceToDevice, stream);
    {
        int n = BM_ * D_;
        k_cvt<<<(n + 255) / 256, 256, 0, stream>>>(att_feats, xb, n);
    }

    for (int i = 0; i < L_; ++i) {
        const size_t wsz = (size_t)D_ * D_;
        int nw = D_ * D_;
        k_cvt<<<(nw + 255) / 256, 256, 0, stream>>>(Wq2 + i * wsz, Wb + 0 * wsz, nw);
        k_cvt<<<(nw + 255) / 256, 256, 0, stream>>>(Wk  + i * wsz, Wb + 1 * wsz, nw);
        k_cvt<<<(nw + 255) / 256, 256, 0, stream>>>(Wv1 + i * wsz, Wb + 2 * wsz, nw);
        k_cvt<<<(nw + 255) / 256, 256, 0, stream>>>(Wv2 + i * wsz, Wb + 3 * wsz, nw);
        int nc = HD_ * HD_;
        k_cvt<<<(nc + 255) / 256, 256, 0, stream>>>(Wc + (size_t)i * nc, Wcb, nc);

        ProjPtrs pp;
        pp.bias[0] = bq2 + i * D_;  pp.bias[1] = bk + i * D_;
        pp.bias[2] = bv1 + i * D_;  pp.bias[3] = bv2 + i * D_;
        pp.gw[0] = gq2w + i * D_;   pp.gw[1] = gkw + i * D_;
        pp.gw[2] = gv1w + i * D_;   pp.gw[3] = gv2w + i * D_;
        pp.gb[0] = gq2b + i * D_;   pp.gb[1] = gkb + i * D_;
        pp.gb[2] = gv1b + i * D_;   pp.gb[3] = gv2b + i * D_;
        pp.out[0] = q2b; pp.out[1] = kb; pp.out[2] = v1b; pp.out[3] = v2b;
        pp.v2T = v2T;
        k_proj<<<dim3(BM_ / 64, H_, 4), 256, 0, stream>>>(xb, Wb, pp);

        int nvk = (int)projElems;
        k_mul_bf16<<<(nvk + 255) / 256, 256, 0, stream>>>(v1b, kb, v1k, nvk);

        k_scores<<<dim3(13, BH_), 256, 0, stream>>>(q2b, kb, att_mask, alpha);
        k_attn<<<dim3(13, BH_), 256, 0, stream>>>(alpha, v2T, v1k, Wcb, bc + (size_t)i * HD_, gattn);
        k_resln<<<BM_, 256, 0, stream>>>(x, gattn, lnw + i * D_, lnb + i * D_, xb,
                                         (i == L_ - 1) ? x_out : nullptr);
    }

    int np = B_ * D_;
    k_pool<<<(np + 255) / 256, 256, 0, stream>>>(x, att_mask, gv_out);
}
